// MultiHeadAttention_31988916421142
// MI455X (gfx1250) — matrix operand order repaired and checked by reading, not compiled, not run on hardware
//
#include <hip/hip_runtime.h>
#include <hip/hip_bf16.h>
#include <stdint.h>

// ---------------------------------------------------------------------------
// Local causal attention (window=256, look_backward=1) + output projection
// for MI455X (gfx1250, wave32, WMMA + async-to-LDS).
//
// HBM-bound (~140MB vs ~34 GFLOP => ~6us floor @ 23.3TB/s), so:
//  - K/V/W converted once to f16 in scratch (coalesced streaming pass);
//    f16 K/V (33.5MB) stay L2-resident across overlapping windows
//  - attention stages K and V windows into LDS with
//    GLOBAL_LOAD_ASYNC_TO_LDS_B128 (ASYNCcnt; no VGPR round-trip),
//    V repacked LDS->LDS into transposed form for WMMA B fragments
//  - all matmuls: v_wmma_f32_16x16x32_f16 (f16 in, f32 accumulate)
//  - attention output kept f16 in scratch (native WMMA input for the GEMM)
// ---------------------------------------------------------------------------

typedef _Float16 v16h __attribute__((ext_vector_type(16)));
typedef _Float16 v8h  __attribute__((ext_vector_type(8)));
typedef _Float16 v4h  __attribute__((ext_vector_type(4)));
typedef float    v8f  __attribute__((ext_vector_type(8)));
typedef float    v4f  __attribute__((ext_vector_type(4)));

#define B_  4
#define N_  2048
#define L_  1024
#define H_  16
#define D_  64
#define W_  256
#define NB_ (N_ / W_)

#define NEG_BIG (-3.0e38f)

// LDS layout (bytes) for the attention kernel
#define KS_OFF   0         // [512][64] f16 K window
#define VT_OFF   65536     // [64][512] f16 V^T window
#define VROW_OFF 131072    // [512][64] f16 V window (async landing pad)
#define P_OFF    196608    // [16][16*32] f16 per-wave P repack scratch
#define LDS_BYTES 212992   // < 320KB/WGP

// ---- CDNA5 async global->LDS copy (tracked by ASYNCcnt) -------------------
__device__ __forceinline__ void async_copy_b128(unsigned lds_off,
                                                const void* gptr) {
  asm volatile("global_load_async_to_lds_b128 %0, %1, off"
               :
               : "v"(lds_off), "v"((unsigned long long)(size_t)gptr)
               : "memory");
}
__device__ __forceinline__ void async_wait0() {
#if __has_builtin(__builtin_amdgcn_s_wait_asynccnt)
  __builtin_amdgcn_s_wait_asynccnt(0);
#else
  asm volatile("s_wait_asynccnt 0x0" ::: "memory");
#endif
}

// wave32: C/D 16x16 layout puts each matrix row across one 16-lane half,
// so row reductions are xor-shuffles with masks 1,2,4,8 (stay within half).
__device__ __forceinline__ float half_row_max(float x) {
  x = fmaxf(x, __shfl_xor(x, 1, 32));
  x = fmaxf(x, __shfl_xor(x, 2, 32));
  x = fmaxf(x, __shfl_xor(x, 4, 32));
  x = fmaxf(x, __shfl_xor(x, 8, 32));
  return x;
}
__device__ __forceinline__ float half_row_sum(float x) {
  x += __shfl_xor(x, 1, 32);
  x += __shfl_xor(x, 2, 32);
  x += __shfl_xor(x, 4, 32);
  x += __shfl_xor(x, 8, 32);
  return x;
}

__device__ __forceinline__ v8f wmma_f16(v16h a, v16h b, v8f c) {
  // (neg_a, A, neg_b, B, c_mod, C, reuse_a, reuse_b)
  return __builtin_amdgcn_wmma_f32_16x16x32_f16(false, a, false, b,
                                                (short)0, c, false, false);
}

// ---------------------------------------------------------------------------
// Kernel 0: elementwise f32 -> f16 (used for W_out, K, V)
// ---------------------------------------------------------------------------
__global__ void ConvertF16_kernel(const float* __restrict__ src,
                                  _Float16* __restrict__ dst, int total) {
  int i = (blockIdx.x * blockDim.x + threadIdx.x) * 4;
  const int stride = gridDim.x * blockDim.x * 4;
  for (; i < total; i += stride) {
    v4f x = *(const v4f*)(src + i);
    v4h y;
    y[0] = (_Float16)x[0]; y[1] = (_Float16)x[1];
    y[2] = (_Float16)x[2]; y[3] = (_Float16)x[3];
    *(v4h*)(dst + i) = y;
  }
}

// ---------------------------------------------------------------------------
// Kernel 1: flash-style local attention.
// grid = (NB, H, B), block = 512 (16 waves); wave w handles query tile w.
// K/V windows async-copied into LDS; V repacked to V^T for B fragments.
// ---------------------------------------------------------------------------
__global__ __launch_bounds__(512) void LocalAttn_kernel(
    const float* __restrict__ q, const _Float16* __restrict__ Kh,
    const _Float16* __restrict__ Vh, _Float16* __restrict__ X) {
  extern __shared__ char smem[];
  _Float16* Ks   = (_Float16*)(smem + KS_OFF);    // [512][64]
  _Float16* Vt   = (_Float16*)(smem + VT_OFF);    // [64][512]
  _Float16* Vrow = (_Float16*)(smem + VROW_OFF);  // [512][64]
  _Float16* Pls  = (_Float16*)(smem + P_OFF);     // [16][16*32]

  const int bucket = blockIdx.x, h = blockIdx.y, b = blockIdx.z;
  const int tid  = threadIdx.x;
  const int lane = tid & 31, wave = tid >> 5;
  const int half = lane >> 4, lm = lane & 15;

  // ---- async-stage K window rows and V window rows (f16, 128B/row) --------
  const int  j     = tid;                     // one key row per thread
  const bool jval  = (bucket > 0) || (j >= W_);  // bucket 0: j<W_ is pad
  if (jval) {                                  // wave-uniform guard
    const size_t rowoff = ((size_t)(b * N_ + (bucket - 1) * W_ + j)) * L_ +
                          (size_t)h * D_;
    const char* kr = (const char*)(Kh + rowoff);
    const char* vr = (const char*)(Vh + rowoff);
    const unsigned koff = (unsigned)(KS_OFF   + j * (D_ * 2));
    const unsigned voff = (unsigned)(VROW_OFF + j * (D_ * 2));
#pragma unroll
    for (int s = 0; s < 8; ++s) {             // 64 halves = 8 x 16B
      async_copy_b128(koff + s * 16, kr + s * 16);
      async_copy_b128(voff + s * 16, vr + s * 16);
    }
  }
  async_wait0();                              // own async writes visible
  // repack own V row into transposed V^T (LDS->LDS, no barrier needed yet)
  if (jval) {
#pragma unroll
    for (int d8 = 0; d8 < D_; d8 += 8) {
      v8h x = *(const v8h*)(Vrow + j * D_ + d8);
#pragma unroll
      for (int t = 0; t < 8; ++t) Vt[(d8 + t) * 512 + j] = x[t];
    }
  }
  __syncthreads();                            // publish Ks + Vt to all waves

  const int   qt    = wave;                   // query tile (16 rows)
  const int   qrow  = bucket * W_ + qt * 16;
  const float scale = 0.125f * 1.44269504088896340736f; // d^-0.5 * log2(e)

  // ---- Q fragments (A layout, ISA 7.12.2): two K=32 chunks ----------------
  // elems 0..7  <- kdim f*32 +      half*8 + (0..7)
  // elems 8..15 <- kdim f*32 + 16 + half*8 + (0..7)
  v16h qa[2];
  {
    const float* qr = q + ((size_t)(b * N_ + qrow + lm)) * L_ + h * D_;
#pragma unroll
    for (int f = 0; f < 2; ++f) {
#pragma unroll
      for (int r = 0; r < 2; ++r) {
        const int kd = f * 32 + r * 16 + half * 8;
        v4f lo = *(const v4f*)(qr + kd);
        v4f hi = *(const v4f*)(qr + kd + 4);
#pragma unroll
        for (int t = 0; t < 4; ++t) {
          qa[f][r * 8 + t]     = (_Float16)(lo[t] * scale);
          qa[f][r * 8 + 4 + t] = (_Float16)(hi[t] * scale);
        }
      }
    }
  }

  v8f acc[4] = {};                // O tile 16x64 (4 N-tiles), C layout
  float mrow[8], lrow[8];         // per-row running max / sum (row = v+half*8)
#pragma unroll
  for (int i = 0; i < 8; ++i) { mrow[i] = NEG_BIG; lrow[i] = 0.f; }

  _Float16* Pw = Pls + wave * (16 * 32);

  // key chunks of 32: c=0..7 prev bucket (skip for bucket 0), 8..15 current.
  const int c0   = (bucket == 0) ? 8 : 0;
  const int cEnd = 8 + (qt * 16 + 15) / 32;   // last chunk touching diagonal

  for (int c = c0; c <= cEnd; ++c) {
    // ---- S = Q @ K^T : two 16x16 tiles, K-dim 64 split as 2x32 -----------
    v8f S[2];
#pragma unroll
    for (int t = 0; t < 2; ++t) {
      const _Float16* kr = Ks + (c * 32 + t * 16 + lm) * D_;  // lane = key col
      v16h b0, b1;
      {
        v8h x0 = *(const v8h*)(kr + half * 8);
        v8h x1 = *(const v8h*)(kr + 16 + half * 8);
        v8h y0 = *(const v8h*)(kr + 32 + half * 8);
        v8h y1 = *(const v8h*)(kr + 48 + half * 8);
#pragma unroll
        for (int e = 0; e < 8; ++e) {
          b0[e] = x0[e]; b0[8 + e] = x1[e];
          b1[e] = y0[e]; b1[8 + e] = y1[e];
        }
      }
      v8f z = {};
      z = wmma_f16(qa[0], b0, z);
      z = wmma_f16(qa[1], b1, z);
      S[t] = z;
    }

    // ---- causal mask (only chunks in the current bucket) -----------------
    if (c >= 8) {
#pragma unroll
      for (int t = 0; t < 2; ++t) {
        const int kl = c * 32 + t * 16 + lm - W_;   // key pos within bucket
#pragma unroll
        for (int vv = 0; vv < 8; ++vv) {
          const int ql = qt * 16 + vv + half * 8;   // query pos within bucket
          if (kl > ql) S[t][vv] = NEG_BIG;
        }
      }
    }

    // ---- online softmax (base-2) -----------------------------------------
    float alpha[8];
#pragma unroll
    for (int vv = 0; vv < 8; ++vv) {
      float rm = half_row_max(fmaxf(S[0][vv], S[1][vv]));
      float mn = fmaxf(mrow[vv], rm);
      alpha[vv] = exp2f(mrow[vv] - mn);
      mrow[vv]  = mn;
      S[0][vv]  = exp2f(S[0][vv] - mn);
      S[1][vv]  = exp2f(S[1][vv] - mn);
      lrow[vv]  = lrow[vv] * alpha[vv] + half_row_sum(S[0][vv] + S[1][vv]);
    }
#pragma unroll
    for (int nc = 0; nc < 4; ++nc)
#pragma unroll
      for (int vv = 0; vv < 8; ++vv) acc[nc][vv] *= alpha[vv];

    // ---- repack P (C layout) -> A layout via per-wave LDS scratch --------
#pragma unroll
    for (int t = 0; t < 2; ++t)
#pragma unroll
      for (int vv = 0; vv < 8; ++vv)
        Pw[(vv + half * 8) * 32 + t * 16 + lm] = (_Float16)S[t][vv];
    // same-wave LDS ops are in-order (DScnt); no barrier needed.
    v16h pa;
    {
      const _Float16* pr = Pw + lm * 32;
      v8h a0 = *(const v8h*)(pr + half * 8);
      v8h a1 = *(const v8h*)(pr + 16 + half * 8);
#pragma unroll
      for (int e = 0; e < 8; ++e) { pa[e] = a0[e]; pa[8 + e] = a1[e]; }
    }

    // ---- O += P @ V  (B fragments from transposed V: contiguous reads) ---
#pragma unroll
    for (int nc = 0; nc < 4; ++nc) {
      const _Float16* vr = Vt + (nc * 16 + lm) * 512 + c * 32 + half * 8;
      v8h b0 = *(const v8h*)(vr);
      v8h b1 = *(const v8h*)(vr + 16);
      v16h vb;
#pragma unroll
      for (int e = 0; e < 8; ++e) { vb[e] = b0[e]; vb[8 + e] = b1[e]; }
      acc[nc] = wmma_f16(pa, vb, acc[nc]);
    }
  }

  // ---- normalize and store attention output as f16 ------------------------
#pragma unroll
  for (int vv = 0; vv < 8; ++vv) {
    const float inv = 1.0f / lrow[vv];
    const int   row = qrow + vv + half * 8;
    _Float16* xr = X + ((size_t)(b * N_ + row)) * L_ + h * D_;
#pragma unroll
    for (int nc = 0; nc < 4; ++nc)
      xr[nc * 16 + lm] = (_Float16)(acc[nc][vv] * inv);
  }
}

// ---------------------------------------------------------------------------
// Kernel 2: out = X(8192x1024 f16) @ W^T(f16) + b, f32 accumulate.
// grid = (L/128, M/64), block 256 (8 waves as 2x4); wave tile 32x32.
// C[m][n] = sum_k X[m][k] * W[n][k]  -> B fragment reads W rows contiguously.
// ---------------------------------------------------------------------------
__global__ __launch_bounds__(256) void OutProj_kernel(
    const _Float16* __restrict__ X, const _Float16* __restrict__ Wh,
    const float* __restrict__ bias, float* __restrict__ out) {
  const int tid  = threadIdx.x;
  const int lane = tid & 31, wave = tid >> 5;
  const int half = lane >> 4, lm = lane & 15;
  const int wm = wave >> 2, wn = wave & 3;
  const int mbase = blockIdx.y * 64 + wm * 32;
  const int nbase = blockIdx.x * 128 + wn * 32;

  v8f acc[2][2] = {};
  for (int kb = 0; kb < L_; kb += 32) {
    v16h a[2], bf[2];
#pragma unroll
    for (int mi = 0; mi < 2; ++mi) {
      const _Float16* xr = X + (size_t)(mbase + mi * 16 + lm) * L_ + kb;
      v8h x0 = *(const v8h*)(xr + half * 8);
      v8h x1 = *(const v8h*)(xr + 16 + half * 8);
#pragma unroll
      for (int e = 0; e < 8; ++e) { a[mi][e] = x0[e]; a[mi][8 + e] = x1[e]; }
    }
#pragma unroll
    for (int ni = 0; ni < 2; ++ni) {
      const _Float16* wr =
          Wh + (size_t)(nbase + ni * 16 + lm) * L_ + kb + half * 8;
      v8h w0 = *(const v8h*)(wr);
      v8h w1 = *(const v8h*)(wr + 16);
#pragma unroll
      for (int e = 0; e < 8; ++e) { bf[ni][e] = w0[e]; bf[ni][8 + e] = w1[e]; }
    }
#pragma unroll
    for (int mi = 0; mi < 2; ++mi)
#pragma unroll
      for (int ni = 0; ni < 2; ++ni)
        acc[mi][ni] = wmma_f16(a[mi], bf[ni], acc[mi][ni]);
  }

#pragma unroll
  for (int mi = 0; mi < 2; ++mi)
#pragma unroll
    for (int ni = 0; ni < 2; ++ni) {
      const int col = nbase + ni * 16 + lm;
      const float bb = bias[col];
#pragma unroll
      for (int vv = 0; vv < 8; ++vv) {
        const int row = mbase + mi * 16 + vv + half * 8;
        out[(size_t)row * L_ + col] = acc[mi][ni][vv] + bb;
      }
    }
}

// ---------------------------------------------------------------------------
extern "C" void kernel_launch(void* const* d_in, const int* in_sizes, int n_in,
                              void* d_out, int out_size, void* d_ws,
                              size_t ws_size, hipStream_t stream) {
  (void)in_sizes; (void)n_in; (void)out_size; (void)ws_size;
  const float* q    = (const float*)d_in[0];
  const float* k    = (const float*)d_in[1];
  const float* v    = (const float*)d_in[2];
  const float* W    = (const float*)d_in[3];
  const float* bias = (const float*)d_in[4];
  float* out = (float*)d_out;

  // scratch: X f16 (16MB) | W f16 (2MB) | K f16 (16MB) | V f16 (16MB)
  const size_t QKV_ELEMS = (size_t)B_ * N_ * L_;
  _Float16* X  = (_Float16*)d_ws;
  _Float16* Wh = X + QKV_ELEMS;
  _Float16* Kh = Wh + (size_t)L_ * L_;
  _Float16* Vh = Kh + QKV_ELEMS;

  ConvertF16_kernel<<<1024, 256, 0, stream>>>(W, Wh, L_ * L_);
  ConvertF16_kernel<<<4096, 256, 0, stream>>>(k, Kh, (int)QKV_ELEMS);
  ConvertF16_kernel<<<4096, 256, 0, stream>>>(v, Vh, (int)QKV_ELEMS);

  LocalAttn_kernel<<<dim3(NB_, H_, B_), 512, LDS_BYTES, stream>>>(q, Kh, Vh,
                                                                  X);

  OutProj_kernel<<<dim3(L_ / 128, (B_ * N_) / 64), 256, 0, stream>>>(X, Wh,
                                                                     bias, out);
}
